// EHEM_26027501814330
// MI455X (gfx1250) — matrix operand order repaired and checked by reading, not compiled, not run on hardware
//
#include <hip/hip_runtime.h>
#include <hip/hip_bf16.h>
#include <math.h>

// ============================================================================
// EHEM forward for gfx1250 (MI455X). All dense contractions use
// v_wmma_f32_16x16x32_f16 (f16 inputs, fp32 accumulate). DGCNN edge-conv is a
// fused gather-GEMM + BN-stats + max-over-k (keeps the 671MB graph-feature
// tensor out of HBM). Attention core / softmax / LN / BN finalize are fp32.
//
// Input index map (setup_inputs() dict insertion order, recursive):
//   0: src (2048,8,7,3)
//   dgcnn: c1.w=1 g=2 be=3 | c2: 4,5,6 | c3: 7,8,9 | c4: 10,11,12
//          c5: 13,14,15 | c6: 16,17,18
//   tlayer blocks (16 leaves each: wq,bq,wk,bk,wv,bv,wo,bo,g1,b1,w1,fb1,w2,fb2,g2,b2):
//          hsa1=19..34  hsa2=35..50  hca1=51..66  hca2=67..82
//   nm1: 83,84  nm2: 85,86  nm3: 87,88
//   dec0: 89,90  dec1: 91,92  dec3: 93,94
//   tr1: 95,96  tr2: 97,98  dec2: 99,100  emb: 101,102
// ============================================================================

typedef __attribute__((ext_vector_type(16))) _Float16 v16h;
typedef __attribute__((ext_vector_type(8)))  float    v8f;

#define LEAKY 0.2f
#define BN_EPS 1e-5f

// ---- workspace layout (float offsets). Total = F_END*4 bytes (~386 MB) ----
static const size_t F_TK0 = 0;           // 8,388,608 token buf (16384x512)
static const size_t F_TK1 = 8388608;     // 8,388,608
static const size_t F_SQ  = 16777216;    // 8,388,608 tlayer scratch
static const size_t F_SK  = 25165824;    // 8,388,608
static const size_t F_SV  = 33554432;    // 8,388,608
static const size_t F_SO  = 41943040;    // 8,388,608
static const size_t F_BIG = 50331648;    // 12,582,912
static const size_t F_AUX = 62914560;    // 12,582,912
static const size_t F_XI1 = 75497472;    // 2,088,960
static const size_t F_XI2 = 77586432;    // 2,088,960
static const size_t F_FIA2= 79675392;    // 4,194,304
static const size_t F_F2  = 83869696;    // 4,194,304
static const size_t F_Q2  = 88064000;    // 2,097,152
static const size_t F_KV2 = 90161152;    // 2,097,152
static const size_t F_FI1 = 92258304;    // 4,194,304
static const size_t F_END = 96452608;
// DGCNN phase (overlaps SQ/SK/SV before the transformer phase starts):
static const size_t F_X0   = F_SQ;            // 294,912
static const size_t F_IDX  = F_SQ + 300000;   // 327,680 ints
static const size_t F_STAT = F_SQ + 700000;   // chanS[512], chanS2[512]
static const size_t F_YMAX = F_SK;            // 4,194,304
static const size_t F_XCAT = F_SV;            // 8,388,608

// ---- monotonic float<->uint encode for atomicMax on floats ----
__device__ __forceinline__ unsigned fenc(float f) {
  unsigned b = __float_as_uint(f);
  return (b & 0x80000000u) ? ~b : (b | 0x80000000u);
}
__device__ __forceinline__ float fdec(unsigned u) {
  return (u & 0x80000000u) ? __uint_as_float(u & 0x7fffffffu)
                           : __uint_as_float(~u);
}
#define FENC_NEGINF 0x007FFFFFu   // fenc(-inf)

// pack two f32 -> packed f16x2 (RNE via scalar casts)
__device__ __forceinline__ unsigned pack2(float a, float b) {
  unsigned short ha = __builtin_bit_cast(unsigned short, (_Float16)a);
  unsigned short hb = __builtin_bit_cast(unsigned short, (_Float16)b);
  return (unsigned)ha | ((unsigned)hb << 16);
}

// ============================================================================
// LDS-tiled WMMA GEMM: C[M,N] = act(A[M,K] * B[K,N] + bias)
// A row-major lda; B row-major ldb (bT=0) or B is (N,K) row-major (bT=1).
// 256 threads = 8 waves, 64x64 C-tile; each wave does a 16x32 strip
// (2 WMMAs sharing one A fragment). Interior tiles take unguarded,
// b128-vectorized (alignment permitting) load paths + next-chunk prefetch.
// ============================================================================
__global__ void __launch_bounds__(256)
k_gemm(const float* __restrict__ A, int lda,
       const float* __restrict__ B, int ldb, int bT,
       const float* __restrict__ bias, float* __restrict__ C, int ldc,
       int M, int N, int K, int act)
{
  __shared__ _Float16 As[64][36];   // [m][k]
  __shared__ _Float16 Bs[64][36];   // transposed: [n][k]
  int tid  = threadIdx.x;
  int lane = tid & 31, wid = tid >> 5;
  int m0 = blockIdx.y * 64, n0 = blockIdx.x * 64;
  int wr = (wid >> 1) * 16;   // wave row base: 0/16/32/48
  int wc = (wid & 1) * 32;    // wave col base: 0/32
  int half = lane >> 4, l15 = lane & 15;
  bool mFull = (m0 + 64 <= M);
  bool nFull = (n0 + 64 <= N);
  bool a4 = ((lda & 3) == 0);
  bool b4 = ((ldb & 3) == 0);
  v8f acc0 = {}, acc1 = {};
  int nk = (K + 31) >> 5;
  for (int kc = 0; kc < nk; ++kc) {
    int k0 = kc * 32;
    bool kFull = (k0 + 32 <= K);
    bool pf = (kc + 1 < nk);
    // ---- stage A (64 x 32) ----
    if (mFull && kFull) {
      if (a4) {
#pragma unroll
        for (int i4 = tid * 4; i4 < 2048; i4 += 1024) {   // 2 iters, b128 loads
          int r = i4 >> 5, c = i4 & 31;
          const float* ap = A + (size_t)(m0 + r) * lda + (k0 + c);
          float4 f = *(const float4*)ap;
          unsigned lo = pack2(f.x, f.y), hi = pack2(f.z, f.w);
          *(unsigned long long*)&As[r][c] =
              (unsigned long long)lo | ((unsigned long long)hi << 32);
          if (pf) __builtin_prefetch(ap + 32, 0, 3);
        }
      } else {
#pragma unroll
        for (int i2 = tid * 2; i2 < 2048; i2 += 512) {
          int r = i2 >> 5, c = i2 & 31;
          const float* ap = A + (size_t)(m0 + r) * lda + (k0 + c);
          *(unsigned*)&As[r][c] = pack2(ap[0], ap[1]);
        }
      }
    } else {
      for (int i = tid; i < 2048; i += 256) {
        int r = i >> 5, c = i & 31;
        int gm = m0 + r, gk = k0 + c;
        As[r][c] = (_Float16)((gm < M && gk < K) ? A[(size_t)gm * lda + gk] : 0.0f);
      }
    }
    // ---- stage B transposed (Bs[n][k], 64 x 32) ----
    if (nFull && kFull) {
      if (bT) {
        if (b4) {
#pragma unroll
          for (int i4 = tid * 4; i4 < 2048; i4 += 1024) {
            int n = i4 >> 5, c = i4 & 31;
            const float* bp = B + (size_t)(n0 + n) * ldb + (k0 + c);
            float4 f = *(const float4*)bp;
            unsigned lo = pack2(f.x, f.y), hi = pack2(f.z, f.w);
            *(unsigned long long*)&Bs[n][c] =
                (unsigned long long)lo | ((unsigned long long)hi << 32);
            if (pf) __builtin_prefetch(bp + 32, 0, 3);
          }
        } else {
#pragma unroll
          for (int i2 = tid * 2; i2 < 2048; i2 += 512) {
            int n = i2 >> 5, c = i2 & 31;
            const float* bp = B + (size_t)(n0 + n) * ldb + (k0 + c);
            *(unsigned*)&Bs[n][c] = pack2(bp[0], bp[1]);
          }
        }
      } else {
        if (b4) {
          // one pass: thread -> (k-pair kp, 4 columns n4)
          int kp = tid >> 4, n4 = (tid & 15) * 4;
          const float* bp = B + (size_t)(k0 + kp * 2) * ldb + (n0 + n4);
          float4 r0 = *(const float4*)bp;
          float4 r1 = *(const float4*)(bp + ldb);
          *(unsigned*)&Bs[n4 + 0][kp * 2] = pack2(r0.x, r1.x);
          *(unsigned*)&Bs[n4 + 1][kp * 2] = pack2(r0.y, r1.y);
          *(unsigned*)&Bs[n4 + 2][kp * 2] = pack2(r0.z, r1.z);
          *(unsigned*)&Bs[n4 + 3][kp * 2] = pack2(r0.w, r1.w);
          if (pf) __builtin_prefetch(bp + (size_t)32 * ldb, 0, 3);
        } else {
#pragma unroll
          for (int i = tid; i < 1024; i += 256) {
            int n = i & 63, kp = i >> 6;
            const float* bp = B + (size_t)(k0 + kp * 2) * ldb + (n0 + n);
            *(unsigned*)&Bs[n][kp * 2] = pack2(bp[0], bp[ldb]);
          }
        }
      }
    } else {
      for (int i = tid; i < 2048; i += 256) {
        int n = i >> 5, c = i & 31;
        int gk = k0 + c, gn = n0 + n;
        float v = 0.0f;
        if (gk < K && gn < N)
          v = bT ? B[(size_t)gn * ldb + gk] : B[(size_t)gk * ldb + gn];
        Bs[n][c] = (_Float16)v;
      }
    }
    __syncthreads();
    // ---- fragments (paired b32 LDS reads, contiguous k = 16*half + s in A and B) ----
    union { v16h v; unsigned u[8]; } Af, Bf0, Bf1;
    const unsigned* arow = (const unsigned*)&As[wr + l15][0];
    int oA0 = half ? 8 : 0, oA1 = half ? 8 : 0;
#pragma unroll
    for (int p = 0; p < 4; ++p) Af.u[p] = arow[p + oA0];
#pragma unroll
    for (int p = 4; p < 8; ++p) Af.u[p] = arow[p + oA1];
    const unsigned* brow0 = (const unsigned*)&Bs[wc + l15][0];
    const unsigned* brow1 = (const unsigned*)&Bs[wc + 16 + l15][0];
    int oB = half ? 8 : 0;
#pragma unroll
    for (int p = 0; p < 8; ++p) { Bf0.u[p] = brow0[p + oB]; Bf1.u[p] = brow1[p + oB]; }
    acc0 = __builtin_amdgcn_wmma_f32_16x16x32_f16(false, Af.v, false, Bf0.v,
                                                  (short)0, acc0, false, false);
    acc1 = __builtin_amdgcn_wmma_f32_16x16x32_f16(false, Af.v, false, Bf1.v,
                                                  (short)0, acc1, false, false);
    __syncthreads();
  }
#pragma unroll
  for (int r = 0; r < 8; ++r) {
    int gm = m0 + wr + r + (half ? 8 : 0);
    if (gm < M) {
      int gn0 = n0 + wc + l15;
      if (gn0 < N) {
        float v = acc0[r];
        if (bias) v += bias[gn0];
        if (act == 1)      v = v > 0.f ? v : 0.f;
        else if (act == 2) v = v > 0.f ? v : LEAKY * v;
        C[(size_t)gm * ldc + gn0] = v;
      }
      int gn1 = gn0 + 16;
      if (gn1 < N) {
        float v = acc1[r];
        if (bias) v += bias[gn1];
        if (act == 1)      v = v > 0.f ? v : 0.f;
        else if (act == 2) v = v > 0.f ? v : LEAKY * v;
        C[(size_t)gm * ldc + gn1] = v;
      }
    }
  }
}

// ============================================================================
// src (2048,8,7,3) -> x0[cloud=512][c=18][n=32]
// ============================================================================
__global__ void k_prep(const float* __restrict__ src, float* __restrict__ x0)
{
  int i = blockIdx.x * 256 + threadIdx.x;
  if (i >= 512 * 18 * 32) return;
  int n = i & 31;
  int c = (i >> 5) % 18;
  int cloud = i / (18 * 32);
  int b = cloud >> 6;
  int t = (cloud & 63) * 32 + n;
  x0[i] = src[((size_t)(t * 8 + b) * 7 + c / 3) * 3 + (c % 3)];
}

// ============================================================================
// KNN (k=20 of 32 points). One wave per cloud. Stable top-k like lax.top_k.
// ============================================================================
__global__ void __launch_bounds__(32)
k_knn(const float* __restrict__ x, int cloudStride, int C, int* __restrict__ idx)
{
  __shared__ float xs[128][33];
  __shared__ float sxx[32];
  int n = threadIdx.x, cloud = blockIdx.x;
  const float* xp = x + (size_t)cloud * cloudStride;
  for (int i = n; i < C * 32; i += 32) xs[i >> 5][i & 31] = xp[i];
  __syncthreads();
  float xx = 0.f;
  for (int c = 0; c < C; ++c) { float v = xs[c][n]; xx += v * v; }
  sxx[n] = xx;
  __syncthreads();
  float d[32];
  for (int m = 0; m < 32; ++m) {
    float dot = 0.f;
    for (int c = 0; c < C; ++c) dot += xs[c][n] * xs[c][m];
    d[m] = 2.0f * dot - xx - sxx[m];
  }
  unsigned used = 0;
  int* op = idx + cloud * 640 + n * 20;
  for (int k = 0; k < 20; ++k) {
    float best = -3.4e38f; int bi = 0;
    for (int m = 0; m < 32; ++m)
      if (!((used >> m) & 1u) && d[m] > best) { best = d[m]; bi = m; }
    used |= (1u << bi);
    op[k] = bi;
  }
}

// ============================================================================
// Fused edge-conv: y[o,(n,k)] = sum_{c<2C} w[o,c]*feat[c], feat=[x[j]-x[n]; x[n]]
// Gather-GEMM via WMMA: weights staged per o-tile in LDS; B-gather classified
// per k-chunk (scalar-uniform): all-neighbor / all-center / mixed.
// Accumulates BN stats + max-over-k in LDS. One workgroup (4 waves) per cloud.
// ============================================================================
__global__ void __launch_bounds__(128)
k_edgeconv(const float* __restrict__ x, int cloudStride, int Cin, int Cout,
           const float* __restrict__ w, const int* __restrict__ idx,
           float* __restrict__ ymax, float* __restrict__ chanS,
           float* __restrict__ chanS2)
{
  __shared__ _Float16 xf[128][33];
  __shared__ _Float16 wt[16][260];
  __shared__ int idxL[640];
  __shared__ unsigned ymx[256 * 32];
  __shared__ float cs[256], cs2[256];
  int tid = threadIdx.x, cloud = blockIdx.x;
  const float* xp = x + (size_t)cloud * cloudStride;
  for (int i = tid; i < Cin * 32; i += 128) xf[i >> 5][i & 31] = (_Float16)xp[i];
  for (int i = tid; i < 640; i += 128) idxL[i] = idx[cloud * 640 + i];
  for (int i = tid; i < Cout * 32; i += 128) ymx[i] = FENC_NEGINF;
  for (int i = tid; i < Cout; i += 128) { cs[i] = 0.f; cs2[i] = 0.f; }
  __syncthreads();
  int lane = tid & 31, wid = tid >> 5;
  int half = lane >> 4, l15 = lane & 15;
  int K = 2 * Cin;
  int nk = (K + 31) >> 5;
  int Kpad = nk * 32;
  int nOT = Cout >> 4;
  int oA0 = half ? 8 : 0, oA1 = half ? 8 : 0;
  for (int ot = 0; ot < nOT; ++ot) {
    int o0 = ot * 16;
    // stage 16 weight rows (f16, zero-padded to Kpad)
    for (int i = tid; i < 16 * Kpad; i += 128) {
      int m = i / Kpad, kk = i % Kpad;
      wt[m][kk] = (_Float16)((kk < K) ? w[(size_t)(o0 + m) * K + kk] : 0.0f);
    }
    __syncthreads();
    for (int ct = wid; ct < 40; ct += 4) {
      int col = ct * 16 + l15;
      int n = col / 20, ks = col % 20;
      int j = idxL[n * 20 + ks];
      v8f acc = {};
      for (int kc = 0; kc < nk; ++kc) {
        union { v16h v; unsigned u[8]; } Af;
        const unsigned* arow = (const unsigned*)&wt[l15][kc * 32];
#pragma unroll
        for (int p = 0; p < 4; ++p) Af.u[p] = arow[p + oA0];
#pragma unroll
        for (int p = 4; p < 8; ++p) Af.u[p] = arow[p + oA1];
        v16h bf;
        int cstart = kc * 32;
        int kb0 = cstart + (half ? 16 : 0);
        if (cstart + 32 <= Cin) {              // all-neighbor chunk
#pragma unroll
          for (int e = 0; e < 16; ++e) {
            int kb = kb0 + e;
            bf[e] = (_Float16)(xf[kb][j] - xf[kb][n]);
          }
        } else if (cstart >= Cin && cstart + 32 <= K) {   // all-center chunk
#pragma unroll
          for (int e = 0; e < 16; ++e) {
            int kb = kb0 + e - Cin;
            bf[e] = xf[kb][n];
          }
        } else {                                // mixed/tail (layer 1 only)
#pragma unroll
          for (int e = 0; e < 16; ++e) {
            int kb = kb0 + e;
            _Float16 bv = (_Float16)0.0f;
            if (kb < Cin)    bv = (_Float16)(xf[kb][j] - xf[kb][n]);
            else if (kb < K) bv = xf[kb - Cin][n];
            bf[e] = bv;
          }
        }
        acc = __builtin_amdgcn_wmma_f32_16x16x32_f16(false, Af.v, false, bf,
                                                     (short)0, acc, false, false);
      }
#pragma unroll
      for (int r = 0; r < 8; ++r) {
        int o = o0 + r + (half ? 8 : 0);
        float v = acc[r];
        atomicAdd(&cs[o], v);
        atomicAdd(&cs2[o], v * v);
        atomicMax(&ymx[o * 32 + n], fenc(v));
      }
    }
    __syncthreads();
  }
  float* yout = ymax + (size_t)cloud * Cout * 32;
  for (int i = tid; i < Cout * 32; i += 128) yout[i] = fdec(ymx[i]);
  for (int i = tid; i < Cout; i += 128) {
    atomicAdd(&chanS[i], cs[i]);
    atomicAdd(&chanS2[i], cs2[i]);
  }
}

// BN (over all b,n,k) + leaky on the k-maxima, scatter into xcat slice.
__global__ void k_bnmax_fin(const float* __restrict__ ymax,
                            const float* __restrict__ cS, const float* __restrict__ cS2,
                            const float* __restrict__ g, const float* __restrict__ be,
                            float* __restrict__ outSlice, int Cout, float invC)
{
  int i = blockIdx.x * 256 + threadIdx.x;
  int total = 512 * Cout * 32;
  if (i >= total) return;
  int n = i & 31;
  int o = (i >> 5) % Cout;
  int cloud = i / (Cout * 32);
  float mean = cS[o] * invC;
  float var  = cS2[o] * invC - mean * mean;
  float sc   = g[o] * rsqrtf(var + BN_EPS);
  float v = (ymax[i] - mean) * sc + be[o];
  v = v > 0.f ? v : LEAKY * v;
  outSlice[(size_t)cloud * 16384 + o * 32 + n] = v;
}

// column (channel) sums for cbr1 BN: y is (R x 512)
__global__ void __launch_bounds__(256)
k_colstats(const float* __restrict__ y, int R, float* __restrict__ sum,
           float* __restrict__ sumsq)
{
  int c1 = threadIdx.x, c2 = threadIdx.x + 256;
  int r0 = blockIdx.x * 64;
  int rend = r0 + 64; if (rend > R) rend = R;
  float a1 = 0, q1 = 0, a2 = 0, q2 = 0;
  for (int r = r0; r < rend; ++r) {
    const float* row = y + ((size_t)r << 9);
    float v = row[c1]; a1 += v; q1 += v * v;
    float u = row[c2]; a2 += u; q2 += u * u;
  }
  atomicAdd(&sum[c1], a1); atomicAdd(&sumsq[c1], q1);
  atomicAdd(&sum[c2], a2); atomicAdd(&sumsq[c2], q2);
}

__global__ void k_bn_apply(float* __restrict__ y, const float* __restrict__ sum,
                           const float* __restrict__ sumsq,
                           const float* __restrict__ g, const float* __restrict__ be,
                           float invC, int total)
{
  int i = blockIdx.x * 256 + threadIdx.x;
  if (i >= total) return;
  int c = i & 511;
  float mean = sum[c] * invC;
  float var  = sumsq[c] * invC - mean * mean;
  float sc   = g[c] * rsqrtf(var + BN_EPS);
  float v = (y[i] - mean) * sc + be[c];
  y[i] = v > 0.f ? v : LEAKY * v;
}

// ============================================================================
// batched transpose: in (B,R,Cc) -> out (B,Cc,R). 32x32 LDS tiles, 256 thr.
// ============================================================================
__global__ void __launch_bounds__(256)
k_transpose(const float* __restrict__ in, float* __restrict__ out, int R, int Cc)
{
  __shared__ float tile[32][33];
  int b = blockIdx.z;
  size_t base = (size_t)b * R * Cc;
  int r0 = blockIdx.y * 32, c0 = blockIdx.x * 32;
  int tx = threadIdx.x & 31, ty = threadIdx.x >> 5;
  for (int k = 0; k < 32; k += 8) {
    int r = r0 + ty + k, c = c0 + tx;
    tile[ty + k][tx] = (r < R && c < Cc) ? in[base + (size_t)r * Cc + c] : 0.f;
  }
  __syncthreads();
  for (int k = 0; k < 32; k += 8) {
    int c = c0 + ty + k, r = r0 + tx;
    if (r < R && c < Cc) out[base + (size_t)c * R + r] = tile[tx][ty + k];
  }
}

// ============================================================================
// windowed attention (win<=16, H=4, dh=128). grid = nWindows*4, block 128.
// ============================================================================
__global__ void __launch_bounds__(128)
k_attn(const float* __restrict__ Q, const float* __restrict__ K,
       const float* __restrict__ V, float* __restrict__ O, int win)
{
  int wi = blockIdx.x >> 2, h = blockIdx.x & 3;
  int t0 = wi * win, dOff = h * 128;
  int tid = threadIdx.x;
  __shared__ float sc[16][17];
  for (int i = tid; i < win * win; i += 128) {
    int qi = i / win, ki = i % win;
    const float* qp = Q + (((size_t)(t0 + qi)) << 9) + dOff;
    const float* kp = K + (((size_t)(t0 + ki)) << 9) + dOff;
    float s = 0.f;
    for (int d = 0; d < 128; ++d) s += qp[d] * kp[d];
    sc[qi][ki] = s * 0.08838834764831845f;   // 1/sqrt(128)
  }
  __syncthreads();
  if (tid < win) {
    float m = -1e30f;
    for (int k2 = 0; k2 < win; ++k2) m = fmaxf(m, sc[tid][k2]);
    float ssum = 0.f;
    for (int k2 = 0; k2 < win; ++k2) { float e = __expf(sc[tid][k2] - m); sc[tid][k2] = e; ssum += e; }
    float inv = 1.0f / ssum;
    for (int k2 = 0; k2 < win; ++k2) sc[tid][k2] *= inv;
  }
  __syncthreads();
  for (int jj = tid; jj < win * 128; jj += 128) {
    int qi = jj >> 7, d = jj & 127;
    float o = 0.f;
    for (int k2 = 0; k2 < win; ++k2)
      o += sc[qi][k2] * V[(((size_t)(t0 + k2)) << 9) + dOff + d];
    O[(((size_t)(t0 + qi)) << 9) + dOff + d] = o;
  }
}

// residual add + LayerNorm over D=512. One block (256 thr) per token.
__global__ void __launch_bounds__(256)
k_add_ln(const float* __restrict__ a, const float* __restrict__ b,
         const float* __restrict__ g, const float* __restrict__ be,
         float* __restrict__ out)
{
  int t = blockIdx.x, tid = threadIdx.x;
  const float* pa = a + ((size_t)t << 9);
  const float* pb = b + ((size_t)t << 9);
  float v0 = pa[tid] + pb[tid];
  float v1 = pa[tid + 256] + pb[tid + 256];
  __shared__ float red[256];
  red[tid] = v0 + v1;
  __syncthreads();
  for (int s2 = 128; s2 > 0; s2 >>= 1) { if (tid < s2) red[tid] += red[tid + s2]; __syncthreads(); }
  float mean = red[0] * (1.0f / 512.0f);
  __syncthreads();
  float d0 = v0 - mean, d1 = v1 - mean;
  red[tid] = d0 * d0 + d1 * d1;
  __syncthreads();
  for (int s2 = 128; s2 > 0; s2 >>= 1) { if (tid < s2) red[tid] += red[tid + s2]; __syncthreads(); }
  float rs = rsqrtf(red[0] * (1.0f / 512.0f) + BN_EPS);
  float* po = out + ((size_t)t << 9);
  po[tid]       = d0 * rs * g[tid] + be[tid];
  po[tid + 256] = d1 * rs * g[tid + 256] + be[tid + 256];
}

// window roll along nw axis: out token (b, w, i) = in token (b, (w+shift)%nw, i)
__global__ void k_roll(const float* __restrict__ in, float* __restrict__ out,
                       int T, int Lb, int win, int srcShift)
{
  int i = blockIdx.x * 256 + threadIdx.x;
  if (i >= T * 512) return;
  int c = i & 511, t = i >> 9;
  int b = t / Lb, l = t % Lb;
  int nw = Lb / win;
  int wI = l / win, ii = l % win;
  int sw = (wI + srcShift) % nw;
  out[i] = in[(((size_t)(b * Lb + sw * win + ii)) << 9) + c];
}

// ---- concat / gather copies ----
__global__ void k_cat48(const float* __restrict__ f1, const float* __restrict__ o2,
                        float* __restrict__ out)
{
  int i = blockIdx.x * 256 + threadIdx.x;
  if (i >= 24576 * 512) return;
  int c = i & 511, row = i >> 9;
  int cloud = row / 48, r = row % 48;
  out[i] = (r < 32) ? f1[((size_t)(cloud * 32 + r) << 9) + c]
                    : o2[((size_t)(cloud * 16 + (r - 32)) << 9) + c];
}
__global__ void k_oddrows(const float* __restrict__ in, float* __restrict__ out)
{
  int i = blockIdx.x * 256 + threadIdx.x;
  if (i >= 8192 * 512) return;
  int c = i & 511, row = i >> 9;
  int b = row >> 10, t = row & 1023;
  out[i] = in[(((size_t)(b * 2048 + 2 * t + 1)) << 9) + c];
}
__global__ void k_cat2(const float* __restrict__ emb, const float* __restrict__ fiahat,
                       float* __restrict__ out)
{
  int i = blockIdx.x * 256 + threadIdx.x;
  if (i >= 16384 * 512) return;
  int c = i & 511, row = i >> 9;
  int b = row >> 11, r = row & 2047;
  if (r < 1024) out[i] = emb[(((size_t)(b * 1024 + r)) << 9) + c];
  else          out[i] = fiahat[(((size_t)(b * 2048 + 2 * (r - 1024))) << 9) + c];
}
__global__ void k_cat2560(const float* __restrict__ q2, const float* __restrict__ f2,
                          const float* __restrict__ fia2, float* __restrict__ out)
{
  int i = blockIdx.x * 256 + threadIdx.x;
  if (i >= 20480 * 512) return;
  int c = i & 511, row = i >> 9;
  int b = row / 2560, r = row % 2560;
  float v;
  if (r < 512)       v = q2[(((size_t)(b * 512 + r)) << 9) + c];
  else if (r < 1536) v = f2[(((size_t)(b * 1024 + r - 512)) << 9) + c];
  else               v = fia2[(((size_t)(b * 1024 + r - 1536)) << 9) + c];
  out[i] = v;
}
__global__ void k_final(const float* __restrict__ xi1, const float* __restrict__ xi2,
                        float* __restrict__ out)
{
  int i = blockIdx.x * 256 + threadIdx.x;
  if (i >= 16384 * 255) return;
  int row = i / 255, c = i % 255;
  int b = row >> 11, t = row & 2047;
  out[i] = (t < 1024) ? xi1[(size_t)(b * 1024 + t) * 255 + c]
                      : xi2[(size_t)(b * 1024 + t - 1024) * 255 + c];
}

// ============================================================================
// host orchestration
// ============================================================================
static inline void gemm(const float* A, int lda, const float* B, int ldb, int bT,
                        const float* bias, float* C, int ldc, int M, int N, int K,
                        int act, hipStream_t s)
{
  dim3 g((N + 63) / 64, (M + 63) / 64, 1);
  k_gemm<<<g, 256, 0, s>>>(A, lda, B, ldb, bT, bias, C, ldc, M, N, K, act);
}

extern "C" void kernel_launch(void* const* d_in, const int* in_sizes, int n_in,
                              void* d_out, int out_size, void* d_ws, size_t ws_size,
                              hipStream_t stream)
{
  float* W = (float*)d_ws;
  auto P = [&](int i) { return (const float*)d_in[i]; };
  const float* SRC = P(0);

  float* TK0 = W + F_TK0; float* TK1 = W + F_TK1;
  float* SQ = W + F_SQ; float* SK = W + F_SK; float* SV = W + F_SV; float* SO = W + F_SO;
  float* BIG = W + F_BIG; float* AUX = W + F_AUX;
  float* XI1 = W + F_XI1; float* XI2 = W + F_XI2;
  float* FIA2 = W + F_FIA2; float* F2b = W + F_F2;
  float* Q2 = W + F_Q2; float* KV2 = W + F_KV2; float* FI1 = W + F_FI1;
  float* x0 = W + F_X0;
  int*   idxb = (int*)(W + F_IDX);
  float* chanS = W + F_STAT; float* chanS2 = W + F_STAT + 512;
  float* ymax = W + F_YMAX; float* xcat = W + F_XCAT;

  // ---------------- DGCNN ----------------
  k_prep<<<(512 * 18 * 32 + 255) / 256, 256, 0, stream>>>(SRC, x0);

  const int cin_[4]  = {18, 64, 64, 128};
  const int cout_[4] = {64, 64, 128, 256};
  const int ico_[4]  = {0, 0, 64, 128};   // input slice offset in xcat (l>0)
  const int oco_[4]  = {0, 64, 128, 256}; // output slice offset in xcat
  const int wb_[4]   = {1, 4, 7, 10};
  for (int l = 0; l < 4; ++l) {
    const float* xin; int cs;
    if (l == 0) { xin = x0; cs = 18 * 32; }
    else        { xin = xcat + ico_[l] * 32; cs = 512 * 32; }
    k_knn<<<512, 32, 0, stream>>>(xin, cs, cin_[l], idxb);
    hipMemsetAsync(chanS, 0, 1024 * sizeof(float), stream);
    k_edgeconv<<<512, 128, 0, stream>>>(xin, cs, cin_[l], cout_[l], P(wb_[l]),
                                        idxb, ymax, chanS, chanS2);
    int total = 512 * cout_[l] * 32;
    k_bnmax_fin<<<(total + 255) / 256, 256, 0, stream>>>(
        ymax, chanS, chanS2, P(wb_[l] + 1), P(wb_[l] + 2),
        xcat + oco_[l] * 32, cout_[l], 1.0f / (512.0f * 32.0f * 20.0f));
  }
  // xcat (512,512ch,32) -> token-major (16384 x 512)
  k_transpose<<<dim3(1, 16, 512), 256, 0, stream>>>(xcat, TK0, 512, 32);
  // c5 (W^T), BN+leaky in place
  gemm(TK0, 512, P(13), 512, 1, nullptr, TK1, 512, 16384, 512, 512, 0, stream);
  hipMemsetAsync(chanS, 0, 1024 * sizeof(float), stream);
  k_colstats<<<256, 256, 0, stream>>>(TK1, 16384, chanS, chanS2);
  k_bn_apply<<<(16384 * 512 + 255) / 256, 256, 0, stream>>>(
      TK1, chanS, chanS2, P(14), P(15), 1.0f / 16384.0f, 16384 * 512);
  // c6
  gemm(TK1, 512, P(16), 512, 1, nullptr, TK0, 512, 16384, 512, 512, 0, stream);
  hipMemsetAsync(chanS, 0, 1024 * sizeof(float), stream);
  k_colstats<<<256, 256, 0, stream>>>(TK0, 16384, chanS, chanS2);
  k_bn_apply<<<(16384 * 512 + 255) / 256, 256, 0, stream>>>(
      TK0, chanS, chanS2, P(17), P(18), 1.0f / 16384.0f, 16384 * 512);
  // TK0 = Fia (16384 x 512)

  // ---------------- transformer helper ----------------
  const float* tp[16];
  auto tl = [&](int base, float* q, float* kv, float* out, int T, int win) {
    for (int i = 0; i < 16; ++i) tp[i] = P(base + i);
    gemm(q,  512, tp[0], 512, 0, tp[1], SQ, 512, T, 512, 512, 0, stream);
    gemm(kv, 512, tp[2], 512, 0, tp[3], SK, 512, T, 512, 512, 0, stream);
    gemm(kv, 512, tp[4], 512, 0, tp[5], SV, 512, T, 512, 512, 0, stream);
    k_attn<<<dim3((T / win) * 4), 128, 0, stream>>>(SQ, SK, SV, SO, win);
    gemm(SO, 512, tp[6], 512, 0, tp[7], SK, 512, T, 512, 512, 0, stream);
    k_add_ln<<<T, 256, 0, stream>>>(q, SK, tp[8], tp[9], SQ);
    gemm(SQ, 512, tp[10], 300, 0, tp[11], SV, 300, T, 512, 300, 1, stream);
    gemm(SV, 300, tp[12], 512, 0, tp[13], SO, 512, T, 300, 512, 0, stream);
    k_add_ln<<<T, 256, 0, stream>>>(SQ, SO, tp[14], tp[15], out);
  };

  // hsa1: L=32, win=16 -> nw=2, roll by 8 % 2 == 0 (identity)
  tl(19, TK0, TK0, TK1, 16384, 16);
  tl(19, TK1, TK1, TK1, 16384, 16);            // f1 = TK1
  // nm1: f1 viewed (8192 x 1024)
  gemm(TK1, 1024, P(83), 512, 0, P(84), TK0, 512, 8192, 1024, 512, 0, stream);
  // hsa2: L=16, nw=1 (identity rolls)
  tl(35, TK0, TK0, TK0, 8192, 16);
  tl(35, TK0, TK0, TK0, 8192, 16);             // o2 = TK0
  // concat [f1 (32 rows), o2 (16 rows)] per cloud -> BIG (24576 x 512)
  k_cat48<<<(24576 * 512 + 255) / 256, 256, 0, stream>>>(TK1, TK0, BIG);
  // dec0 (lin_t, K=48 -> N=32, relu)
  k_transpose<<<dim3(16, 2, 512), 256, 0, stream>>>(BIG, AUX, 48, 512);
  gemm(AUX, 48, P(89), 32, 0, P(90), TK0, 32, 262144, 48, 32, 1, stream);
  k_transpose<<<dim3(1, 16, 512), 256, 0, stream>>>(TK0, TK1, 512, 32);
  // TK1 = Fia_hat (16384 x 512)
  // tr2 (lin_t over t-axis per batch)
  k_transpose<<<dim3(16, 64, 8), 256, 0, stream>>>(TK1, BIG, 2048, 512);
  gemm(BIG, 2048, P(97), 1024, 0, P(98), AUX, 1024, 4096, 2048, 1024, 0, stream);
  k_transpose<<<dim3(32, 16, 8), 256, 0, stream>>>(AUX, TK0, 512, 1024);
  // dec1 -> output_xi1
  gemm(TK0, 512, P(91), 255, 0, P(92), XI1, 255, 8192, 512, 255, 0, stream);
  // emb -> xi1_embed (TK0)
  gemm(XI1, 255, P(101), 512, 0, P(102), TK0, 512, 8192, 255, 512, 0, stream);
  // Fia2 = odd rows of Fia_hat
  k_oddrows<<<(8192 * 512 + 255) / 256, 256, 0, stream>>>(TK1, FIA2);
  // cat [xi1_embed, Fia1] -> BIG (16384 x 512)
  k_cat2<<<(16384 * 512 + 255) / 256, 256, 0, stream>>>(TK0, TK1, BIG);
  // tr1 (relu)
  k_transpose<<<dim3(16, 64, 8), 256, 0, stream>>>(BIG, AUX, 2048, 512);
  gemm(AUX, 2048, P(95), 1024, 0, P(96), BIG, 1024, 4096, 2048, 1024, 1, stream);
  k_transpose<<<dim3(32, 16, 8), 256, 0, stream>>>(BIG, FI1, 512, 1024);
  // hca1: L=1024 per batch, win=16, nw=64, shift 8
  tl(51, FIA2, FI1, TK0, 8192, 16);
  k_roll<<<(8192 * 512 + 255) / 256, 256, 0, stream>>>(TK0, TK1, 8192, 1024, 16, 8);
  tl(51, TK1, TK1, TK1, 8192, 16);
  k_roll<<<(8192 * 512 + 255) / 256, 256, 0, stream>>>(TK1, F2b, 8192, 1024, 16, 56);
  // nm2 / nm3 (views 4096 x 1024)
  gemm(F2b,  1024, P(85), 512, 0, P(86), KV2, 512, 4096, 1024, 512, 0, stream);
  gemm(FIA2, 1024, P(87), 512, 0, P(88), Q2,  512, 4096, 1024, 512, 0, stream);
  // hca2: L=512 per batch, win=8, nw=64, shift 4
  tl(67, Q2, KV2, TK0, 4096, 8);
  k_roll<<<(4096 * 512 + 255) / 256, 256, 0, stream>>>(TK0, TK1, 4096, 512, 8, 4);
  tl(67, TK1, TK1, TK1, 4096, 8);
  k_roll<<<(4096 * 512 + 255) / 256, 256, 0, stream>>>(TK1, Q2, 4096, 512, 8, 60);
  // out2 = [q2, f2, Fia2] -> BIG (20480 x 512)
  k_cat2560<<<(20480 * 512 + 255) / 256, 256, 0, stream>>>(Q2, F2b, FIA2, BIG);
  // dec2 (lin_t, K=2560)
  k_transpose<<<dim3(16, 80, 8), 256, 0, stream>>>(BIG, AUX, 2560, 512);
  gemm(AUX, 2560, P(99), 1024, 0, P(100), BIG, 1024, 4096, 2560, 1024, 0, stream);
  k_transpose<<<dim3(32, 16, 8), 256, 0, stream>>>(BIG, TK0, 512, 1024);
  // dec3 -> output_xi2
  gemm(TK0, 512, P(93), 255, 0, P(94), XI2, 255, 8192, 512, 255, 0, stream);
  // final concat -> d_out (8,2048,255)
  k_final<<<(16384 * 255 + 255) / 256, 256, 0, stream>>>(XI1, XI2, (float*)d_out);
}
